// SinkhornDistance_56719338111039
// MI455X (gfx1250) — matrix operand order repaired and checked by reading, not compiled, not run on hardware
//
#include <hip/hip_runtime.h>

#define REG_EPS 1e-8f
#define N_PTS 4096
#define D_DIM 512
#define INV_N (1.0f / 4096.0f)
#define NEG_INV_REG (-20.0f)   // -1/0.05
#define REG_W 0.05f
#define ROW_CHUNKS 32          // colsum row split (4096/32 = 128 rows/chunk)

typedef __attribute__((ext_vector_type(16))) _Float16 v16h;
typedef __attribute__((ext_vector_type(8)))  float    v8f;

// ---------------------------------------------------------------------------
// 1) Convert one input (4096x512 f32) to f16 and compute per-row |r|^2.
// ---------------------------------------------------------------------------
__global__ void prep_kernel(const float* __restrict__ src,
                            _Float16* __restrict__ dst,
                            float* __restrict__ sq) {
    const int row = blockIdx.x;
    const int t   = threadIdx.x;
    const float* r = src + (size_t)row * D_DIM;
    float a0 = r[2 * t];
    float a1 = r[2 * t + 1];
    _Float16* d = dst + (size_t)row * D_DIM;
    d[2 * t]     = (_Float16)a0;
    d[2 * t + 1] = (_Float16)a1;

    __shared__ float red[256];
    red[t] = a0 * a0 + a1 * a1;
    __syncthreads();
    for (int off = 128; off > 0; off >>= 1) {
        if (t < off) red[t] += red[t + off];
        __syncthreads();
    }
    if (t == 0) sq[row] = red[0];
}

// ---------------------------------------------------------------------------
// 2) WMMA GEMM + fused epilogue (unchanged; emits v_wmma_f32_16x16x32_f16):
//    K = exp(-max(xsq + ysq - 2 x@y^T, 0)/REG), stored f32 (L2 resident).
// ---------------------------------------------------------------------------
__global__ void gemm_expK_kernel(const _Float16* __restrict__ xh,
                                 const _Float16* __restrict__ yh,
                                 const float* __restrict__ xsq,
                                 const float* __restrict__ ysq,
                                 float* __restrict__ Kmat) {
    const int tileM = blockIdx.x * 16;
    const int tileN = blockIdx.y * 16;
    const int lane  = threadIdx.x;          // 0..31 (wave32)
    const int hsel  = lane >> 4;
    const int l     = lane & 15;

    const _Float16* arow = xh + (size_t)(tileM + l) * D_DIM;
    const _Float16* brow = yh + (size_t)(tileN + l) * D_DIM;

    v8f acc = {};
    for (int k0 = 0; k0 < D_DIM; k0 += 32) {
        v16h a, b;
        #pragma unroll
        for (int i = 0; i < 4; ++i) {
            const int ka = k0 + hsel * 8 + 2 * i;
            a[2 * i]         = arow[ka];
            a[2 * i + 1]     = arow[ka + 1];
            a[8 + 2 * i]     = arow[ka + 16];
            a[8 + 2 * i + 1] = arow[ka + 17];
        }
        #pragma unroll
        for (int i = 0; i < 8; ++i) {
            const int kb = k0 + hsel * 8 + 2 * i + (i >= 4 ? 8 : 0);
            b[2 * i]     = brow[kb];
            b[2 * i + 1] = brow[kb + 1];
        }
        acc = __builtin_amdgcn_wmma_f32_16x16x32_f16(
            false, a, false, b, (short)0, acc, false, false);
    }

    const int   n     = tileN + l;
    const float ysq_n = ysq[n];
    #pragma unroll
    for (int r = 0; r < 8; ++r) {
        const int m = tileM + r + hsel * 8;
        float C = xsq[m] + ysq_n - 2.0f * acc[r];
        C = fmaxf(C, 0.0f);
        Kmat[(size_t)m * N_PTS + n] = __expf(C * NEG_INV_REG);
    }
}

// ---------------------------------------------------------------------------
// 3) u,v init
// ---------------------------------------------------------------------------
__global__ void init_uv_kernel(float* __restrict__ u, float* __restrict__ v) {
    const int t = blockIdx.x * 256 + threadIdx.x;
    u[t] = 1.0f;
    v[t] = 1.0f;
}

// ---------------------------------------------------------------------------
// 4a) Column-sum partials for v = nu/(K^T u + eps).
//     Grid (4, ROW_CHUNKS): blockIdx.x = 1024-column group (4 cols/thread,
//     b128 loads), blockIdx.y = 128-row chunk. u chunk staged in LDS.
// ---------------------------------------------------------------------------
__global__ void colsum_partial_kernel(const float* __restrict__ Kmat,
                                      const float* __restrict__ u,
                                      float* __restrict__ part) {
    const int colBase = blockIdx.x * 1024 + threadIdx.x * 4;
    const int rowBase = blockIdx.y * (N_PTS / ROW_CHUNKS);

    __shared__ float us[N_PTS / ROW_CHUNKS];
    if (threadIdx.x < (N_PTS / ROW_CHUNKS)) us[threadIdx.x] = u[rowBase + threadIdx.x];
    __syncthreads();

    float s0 = 0.f, s1 = 0.f, s2 = 0.f, s3 = 0.f;
    #pragma unroll 4
    for (int i = 0; i < (N_PTS / ROW_CHUNKS); ++i) {
        const float4 k = *(const float4*)(Kmat + (size_t)(rowBase + i) * N_PTS + colBase);
        const float ui = us[i];
        s0 = fmaf(k.x, ui, s0);
        s1 = fmaf(k.y, ui, s1);
        s2 = fmaf(k.z, ui, s2);
        s3 = fmaf(k.w, ui, s3);
    }
    *(float4*)(part + (size_t)blockIdx.y * N_PTS + colBase) = make_float4(s0, s1, s2, s3);
}

// 4b) Combine ROW_CHUNKS partials per column -> v
__global__ void colsum_finish_kernel(const float* __restrict__ part,
                                     float* __restrict__ v) {
    const int col = blockIdx.x * 256 + threadIdx.x;
    float s = 0.f;
    #pragma unroll
    for (int c = 0; c < ROW_CHUNKS; ++c) s += part[(size_t)c * N_PTS + col];
    v[col] = INV_N / (s + REG_EPS);
}

// ---------------------------------------------------------------------------
// 5) u = mu/(K v + eps): block-per-row, float4 loads, LDS tree reduce.
// ---------------------------------------------------------------------------
__global__ void rowsum_u_kernel(const float* __restrict__ Kmat,
                                const float* __restrict__ v,
                                float* __restrict__ u) {
    const int row = blockIdx.x;
    const int t   = threadIdx.x;
    const float4* kr = (const float4*)(Kmat + (size_t)row * N_PTS);
    const float4* vv = (const float4*)v;
    float s = 0.f;
    #pragma unroll
    for (int it = 0; it < (N_PTS / 4) / 256; ++it) {
        const int j = it * 256 + t;
        const float4 k = kr[j];
        const float4 w = vv[j];
        s = fmaf(k.x, w.x, s);
        s = fmaf(k.y, w.y, s);
        s = fmaf(k.z, w.z, s);
        s = fmaf(k.w, w.w, s);
    }
    __shared__ float red[256];
    red[t] = s;
    __syncthreads();
    for (int off = 128; off > 0; off >>= 1) {
        if (t < off) red[t] += red[t + off];
        __syncthreads();
    }
    if (t == 0) u[row] = INV_N / (red[0] + REG_EPS);
}

// ---------------------------------------------------------------------------
// 6) Final: sum_{ij} u_i K_ij v_j C_ij with C = -REG*ln(K) (K==0 -> 0).
//    float4 grid-stride, two-stage deterministic reduction.
// ---------------------------------------------------------------------------
__global__ void final_partial_kernel(const float* __restrict__ Kmat,
                                     const float* __restrict__ u,
                                     const float* __restrict__ v,
                                     float* __restrict__ part) {
    const unsigned tid    = blockIdx.x * 256 + threadIdx.x;
    const unsigned stride = gridDim.x * 256;
    const float4* K4 = (const float4*)Kmat;
    const float4* V4 = (const float4*)v;
    const size_t nvec = (size_t)N_PTS * N_PTS / 4;
    float s = 0.f;
    for (size_t q = tid; q < nvec; q += stride) {
        const float4 k = K4[q];
        const size_t idx = q * 4;
        const float  ui  = u[idx >> 12];
        const float4 w   = V4[(idx & (N_PTS - 1)) >> 2];
        if (k.x > 0.f) s = fmaf(ui * w.x, k.x * (-REG_W * __logf(k.x)), s);
        if (k.y > 0.f) s = fmaf(ui * w.y, k.y * (-REG_W * __logf(k.y)), s);
        if (k.z > 0.f) s = fmaf(ui * w.z, k.z * (-REG_W * __logf(k.z)), s);
        if (k.w > 0.f) s = fmaf(ui * w.w, k.w * (-REG_W * __logf(k.w)), s);
    }
    __shared__ float red[256];
    red[threadIdx.x] = s;
    __syncthreads();
    for (int off = 128; off > 0; off >>= 1) {
        if (threadIdx.x < off) red[threadIdx.x] += red[threadIdx.x + off];
        __syncthreads();
    }
    if (threadIdx.x == 0) part[blockIdx.x] = red[0];
}

__global__ void final_reduce_kernel(const float* __restrict__ part,
                                    float* __restrict__ out) {
    __shared__ float red[256];
    red[threadIdx.x] = part[threadIdx.x];
    __syncthreads();
    for (int off = 128; off > 0; off >>= 1) {
        if (threadIdx.x < off) red[threadIdx.x] += red[threadIdx.x + off];
        __syncthreads();
    }
    if (threadIdx.x == 0) out[0] = red[0];
}

// ---------------------------------------------------------------------------
// Launch
// ---------------------------------------------------------------------------
extern "C" void kernel_launch(void* const* d_in, const int* in_sizes, int n_in,
                              void* d_out, int out_size, void* d_ws, size_t ws_size,
                              hipStream_t stream) {
    (void)in_sizes; (void)n_in; (void)out_size; (void)ws_size;
    const float* x = (const float*)d_in[0];
    const float* y = (const float*)d_in[1];
    float* out = (float*)d_out;

    char* ws = (char*)d_ws;
    size_t off = 0;
    _Float16* xh  = (_Float16*)(ws + off); off += (size_t)N_PTS * D_DIM * 2;   // 4 MB
    _Float16* yh  = (_Float16*)(ws + off); off += (size_t)N_PTS * D_DIM * 2;   // 4 MB
    float* xsq    = (float*)(ws + off);    off += (size_t)N_PTS * 4;
    float* ysq    = (float*)(ws + off);    off += (size_t)N_PTS * 4;
    float* u      = (float*)(ws + off);    off += (size_t)N_PTS * 4;
    float* v      = (float*)(ws + off);    off += (size_t)N_PTS * 4;
    float* part   = (float*)(ws + off);    off += 256 * 4;
    off = (off + 255) & ~(size_t)255;
    float* cpart  = (float*)(ws + off);    off += (size_t)ROW_CHUNKS * N_PTS * 4; // 512 KB
    off = (off + 255) & ~(size_t)255;
    float* Kmat   = (float*)(ws + off);    off += (size_t)N_PTS * N_PTS * 4;   // 64 MB, L2-resident

    // 1) f32 -> f16 + row squared norms
    prep_kernel<<<N_PTS, 256, 0, stream>>>(x, xh, xsq);
    prep_kernel<<<N_PTS, 256, 0, stream>>>(y, yh, ysq);

    // 2) WMMA GEMM + exp fusion
    gemm_expK_kernel<<<dim3(N_PTS / 16, N_PTS / 16), 32, 0, stream>>>(
        xh, yh, xsq, ysq, Kmat);

    // 3) Sinkhorn iterations (all L2-resident traffic)
    init_uv_kernel<<<N_PTS / 256, 256, 0, stream>>>(u, v);
    for (int it = 0; it < 100; ++it) {
        colsum_partial_kernel<<<dim3(4, ROW_CHUNKS), 256, 0, stream>>>(Kmat, u, cpart);
        colsum_finish_kernel<<<N_PTS / 256, 256, 0, stream>>>(cpart, v);
        rowsum_u_kernel<<<N_PTS, 256, 0, stream>>>(Kmat, v, u);
    }

    // 4) Final deterministic reduction
    final_partial_kernel<<<256, 256, 0, stream>>>(Kmat, u, v, part);
    final_reduce_kernel<<<1, 256, 0, stream>>>(part, out);
}
